// MllamaSelfAttentionDecoderLayer_23356032155986
// MI455X (gfx1250) — matrix operand order repaired and checked by reading, not compiled, not run on hardware
//
#include <hip/hip_runtime.h>
#include <hip/hip_bf16.h>
#include <math.h>

// ---------------------------------------------------------------------------
// Problem constants (from reference)
// ---------------------------------------------------------------------------
#define BB   4
#define SS   1024
#define HH   4096
#define NQ   32
#define NKV  8
#define DH   128
#define GG   (NQ / NKV)              // 4
#define II   14336
#define QKVD ((NQ + 2 * NKV) * DH)   // 6144
#define EPS  1e-6f
#define MM   (BB * SS)               // GEMM M = 4096 (fixed)

typedef __attribute__((ext_vector_type(16))) _Float16 v16h;
typedef __attribute__((ext_vector_type(8)))  float    v8f;
typedef __attribute__((ext_vector_type(4)))  float    v4f;
typedef __attribute__((ext_vector_type(4)))  _Float16 v4h;

__device__ __forceinline__ v8f wmma_f32_f16(v16h a, v16h b, v8f c) {
    // 8 args: (neg_a, A, neg_b, B, c_mod, C, reuse_a, reuse_b)
    return __builtin_amdgcn_wmma_f32_16x16x32_f16(
        false, a, false, b, (short)0, c, false, false);
}

__device__ __forceinline__ v4h cvt4(v4f f) {
    v4h h;
    #pragma unroll
    for (int j = 0; j < 4; ++j) h[j] = (_Float16)f[j];
    return h;
}

// ---------------------------------------------------------------------------
// Fused residual-add + RMSNorm (float4 vectorized)
// one 256-thread block per (b,s) row of H=4096
// ---------------------------------------------------------------------------
__global__ __launch_bounds__(256)
void rmsnorm_fused_kernel(const float* __restrict__ x,
                          const float* __restrict__ r_in,
                          const float* __restrict__ w,
                          float* __restrict__ h_out,
                          float* __restrict__ r_out) {
    __shared__ float red[256];
    const int row = blockIdx.x;
    const float* xr = x    + (size_t)row * HH;
    const float* rr = r_in + (size_t)row * HH;

    float acc = 0.f;
    #pragma unroll
    for (int i = 0; i < HH / 1024; ++i) {          // 4 iters of 1024 floats
        int c = (threadIdx.x + i * 256) * 4;
        v4f a = *reinterpret_cast<const v4f*>(xr + c);
        v4f b = *reinterpret_cast<const v4f*>(rr + c);
        v4f v = a + b;
        acc += v[0]*v[0] + v[1]*v[1] + v[2]*v[2] + v[3]*v[3];
    }
    red[threadIdx.x] = acc;
    __syncthreads();
    for (int s = 128; s > 0; s >>= 1) {
        if (threadIdx.x < s) red[threadIdx.x] += red[threadIdx.x + s];
        __syncthreads();
    }
    const float inv = rsqrtf(red[0] / (float)HH + EPS);
    #pragma unroll
    for (int i = 0; i < HH / 1024; ++i) {
        int c = (threadIdx.x + i * 256) * 4;
        v4f a = *reinterpret_cast<const v4f*>(xr + c);
        v4f b = *reinterpret_cast<const v4f*>(rr + c);
        v4f g = *reinterpret_cast<const v4f*>(w + c);
        v4f v = a + b;
        *reinterpret_cast<v4f*>(r_out + (size_t)row * HH + c) = v;
        *reinterpret_cast<v4f*>(h_out + (size_t)row * HH + c) = g * v * inv;
    }
}

// ---------------------------------------------------------------------------
// WMMA GEMM:  C[M,N] = A[M,K] * W[K,N], f32 in/out, f16 math, M = 4096.
// N, K compile-time so every address offset folds into immediates.
// 1D grid with an 8(M)-block supertile swizzle: 8 consecutive workgroups
// share one 2MB weight panel -> weight HBM traffic cut 8x via the 192MB L2.
// Register double-buffering hides the next tile's global loads under WMMA.
// ---------------------------------------------------------------------------
template <int N, int K>
__global__ __launch_bounds__(256)
void gemm_f16_wmma(const float* __restrict__ A,
                   const float* __restrict__ W,
                   float* __restrict__ C) {
    __shared__ _Float16 As[128][40];    // [m][k], padded (80B row stride)
    __shared__ _Float16 Bt[128][40];    // [n][k], transposed W tile

    // ---- L2 supertile swizzle (M fixed = 4096 -> 32 m-blocks, GRP=8) ----
    constexpr int NBLK = N / 128;
    const int bid  = blockIdx.x;
    const int grp  = bid / (8 * NBLK);          // m-group of 8
    const int rem  = bid % (8 * NBLK);
    const int m0   = (grp * 8 + (rem & 7)) * 128;
    const int n0   = (rem >> 3) * 128;

    const int tid  = threadIdx.x;
    const int lane = tid & 31;
    const int wave = tid >> 5;
    const int wm   = wave & 3;          // 4 wave-rows of 32
    const int wn   = wave >> 2;         // 2 wave-cols of 64
    const int half = lane >> 4;
    const int ln   = lane & 15;

    // per-thread staging coordinates
    const int arow[4] = { (tid + 0*256) >> 3, (tid + 1*256) >> 3,
                          (tid + 2*256) >> 3, (tid + 3*256) >> 3 };
    const int ac4     = (tid & 7) << 2;
    const int wrow[4] = { (tid + 0*256) >> 5, (tid + 1*256) >> 5,
                          (tid + 2*256) >> 5, (tid + 3*256) >> 5 };
    const int wc4     = (tid & 31) << 2;

    v8f acc[2][4] = {};
    v4f ra[4], rw[4];

    // prologue: load tile k0 = 0 into registers
    #pragma unroll
    for (int i = 0; i < 4; ++i) {
        ra[i] = *reinterpret_cast<const v4f*>(&A[(size_t)(m0 + arow[i]) * K + ac4]);
        rw[i] = *reinterpret_cast<const v4f*>(&W[(size_t)wrow[i] * N + n0 + wc4]);
    }

    for (int k0 = 0; k0 < K; k0 += 32) {
        // ---- commit held registers to LDS (f32 -> f16) ----
        #pragma unroll
        for (int i = 0; i < 4; ++i) {
            *reinterpret_cast<v4h*>(&As[arow[i]][ac4]) = cvt4(ra[i]);
            v4h h = cvt4(rw[i]);
            #pragma unroll
            for (int j = 0; j < 4; ++j) Bt[wc4 + j][wrow[i]] = h[j];
        }
        __syncthreads();

        // ---- issue next tile's global loads (overlap with WMMA) ----
        if (k0 + 32 < K) {
            #pragma unroll
            for (int i = 0; i < 4; ++i) {
                ra[i] = *reinterpret_cast<const v4f*>(
                    &A[(size_t)(m0 + arow[i]) * K + k0 + 32 + ac4]);
                rw[i] = *reinterpret_cast<const v4f*>(
                    &W[(size_t)(k0 + 32 + wrow[i]) * N + n0 + wc4]);
            }
        }

        // ---- fragment loads: contiguous runs => ds_load_b128 pairs ----
        v16h af[2], bf[4];
        #pragma unroll
        for (int mi = 0; mi < 2; ++mi) {
            int row = wm * 32 + mi * 16 + ln;   // A: M = lane%16
            int kb  = half << 4;
            #pragma unroll
            for (int e = 0; e < 8; ++e) af[mi][e]     = As[row][kb + e];
            #pragma unroll
            for (int e = 0; e < 8; ++e) af[mi][8 + e] = As[row][kb + 8 + e];
        }
        #pragma unroll
        for (int nj = 0; nj < 4; ++nj) {
            int col = wn * 64 + nj * 16 + ln;   // B: N = lane%16
            int kb  = half << 4;
            #pragma unroll
            for (int e = 0; e < 16; ++e) bf[nj][e] = Bt[col][kb + e];
        }
        // ---- 8 WMMAs ----
        #pragma unroll
        for (int mi = 0; mi < 2; ++mi)
            #pragma unroll
            for (int nj = 0; nj < 4; ++nj)
                acc[mi][nj] = wmma_f32_f16(af[mi], bf[nj], acc[mi][nj]);
        __syncthreads();
    }

    // ---- store C: one base address per tile, immediate r*N offsets ----
    #pragma unroll
    for (int mi = 0; mi < 2; ++mi)
        #pragma unroll
        for (int nj = 0; nj < 4; ++nj) {
            float* cp = C + (size_t)(m0 + wm * 32 + mi * 16 + (half << 3)) * N
                          + n0 + wn * 64 + nj * 16 + ln;
            #pragma unroll
            for (int r = 0; r < 8; ++r)
                cp[(size_t)r * N] = acc[mi][nj][r];
        }
}

// ---------------------------------------------------------------------------
// RoPE applied in-place to Q (heads 0..31) and K (heads 32..39) inside qkv.
// ---------------------------------------------------------------------------
__global__ void rope_kernel(float* __restrict__ qkv,
                            const float* __restrict__ cosb,
                            const float* __restrict__ sinb) {
    const int total = BB * SS * (NQ + NKV) * (DH / 2);
    for (int idx = blockIdx.x * blockDim.x + threadIdx.x; idx < total;
         idx += gridDim.x * blockDim.x) {
        int d = idx & 63;
        int t = idx >> 6;
        int head = t % (NQ + NKV); t /= (NQ + NKV);
        int s = t % SS;
        int b = t / SS;
        size_t base  = ((size_t)(b * SS + s)) * QKVD + (size_t)head * DH;
        size_t cbase = ((size_t)(b * SS + s)) * DH;
        float x1 = qkv[base + d], x2 = qkv[base + d + 64];
        float c1 = cosb[cbase + d],      s1 = sinb[cbase + d];
        float c2 = cosb[cbase + d + 64], s2 = sinb[cbase + d + 64];
        qkv[base + d]      = x1 * c1 - x2 * s1;   // lo half
        qkv[base + d + 64] = x2 * c2 + x1 * s2;   // hi half
    }
}

// ---------------------------------------------------------------------------
// Flash attention (causal, GQA). One wave per (b, q-head, 16-query block).
// K staged row-major [key][d]; V staged transposed [d][key] so both
// fragment types are contiguous LDS reads. All global loads are float4.
// ---------------------------------------------------------------------------
__global__ __launch_bounds__(32)
void attention_kernel(const float* __restrict__ qkv,
                      float* __restrict__ attn) {
    __shared__ _Float16 Ks[32][136];    // [key][d]   8704 B
    __shared__ _Float16 Vt[128][40];    // [d][key]  10240 B
    __shared__ _Float16 Ps[16][40];     // probs      1280 B

    const int lane  = threadIdx.x;
    const int qblk  = blockIdx.x;
    const int hq    = blockIdx.y;
    const int b     = blockIdx.z;
    const int kv    = hq / GG;
    const int qbase = qblk * 16;
    const int half  = lane >> 4;
    const int ln    = lane & 15;
    const float scale = 0.08838834764831845f;   // 1/sqrt(128)

    // Q fragments: one contiguous 16-float run per dc => global_load_b128 x4
    v16h qf[4];
    {
        const size_t qrow = ((size_t)(b * SS) + qbase + ln) * QKVD + (size_t)hq * DH;
        #pragma unroll
        for (int dc = 0; dc < 4; ++dc) {
            const float* p0 = qkv + qrow + dc * 32 + (half << 4);
            v4f a0 = *reinterpret_cast<const v4f*>(p0);
            v4f a1 = *reinterpret_cast<const v4f*>(p0 + 4);
            v4f b0 = *reinterpret_cast<const v4f*>(p0 + 8);
            v4f b1 = *reinterpret_cast<const v4f*>(p0 + 12);
            #pragma unroll
            for (int j = 0; j < 4; ++j) {
                qf[dc][j]      = (_Float16)(a0[j] * scale);
                qf[dc][4 + j]  = (_Float16)(a1[j] * scale);
                qf[dc][8 + j]  = (_Float16)(b0[j] * scale);
                qf[dc][12 + j] = (_Float16)(b1[j] * scale);
            }
        }
    }

    float m[8], l[8];
    v8f acc[8] = {};
    #pragma unroll
    for (int r = 0; r < 8; ++r) { m[r] = -1e30f; l[r] = 0.f; }

    const size_t kcol = (size_t)NQ * DH + (size_t)kv * DH;
    const size_t vcol = (size_t)(NQ + NKV) * DH + (size_t)kv * DH;

    const int nch = (qbase + 16 + 31) >> 5;     // causal chunk count (32 keys)
    for (int kc = 0; kc < nch; ++kc) {
        // ---- stage K (row-major) and V (transposed) for this chunk ----
        #pragma unroll 4
        for (int i = 0; i < 32; ++i) {
            const size_t srow = ((size_t)(b * SS) + kc * 32 + i) * QKVD;
            v4f fk = *reinterpret_cast<const v4f*>(&qkv[srow + kcol + lane * 4]);
            *reinterpret_cast<v4h*>(&Ks[i][lane * 4]) = cvt4(fk);
            v4f fv = *reinterpret_cast<const v4f*>(&qkv[srow + vcol + lane * 4]);
            v4h hv = cvt4(fv);
            #pragma unroll
            for (int j = 0; j < 4; ++j) Vt[lane * 4 + j][i] = hv[j];
        }
        __syncthreads();

        // ---- scores: two 16x16 tiles over the 32-key chunk ----
        v8f sacc[2] = {};
        #pragma unroll
        for (int t = 0; t < 2; ++t) {
            const int key = t * 16 + ln;        // B-frag: N = key col
            #pragma unroll
            for (int dc = 0; dc < 4; ++dc) {
                v16h kf;
                int db = dc * 32 + (half << 4); // contiguous d run
                #pragma unroll
                for (int e = 0; e < 16; ++e) kf[e] = Ks[key][db + e];
                sacc[t] = wmma_f32_f16(qf[dc], kf, sacc[t]);
            }
        }

        // ---- online softmax update ----
        #pragma unroll
        for (int r = 0; r < 8; ++r) {
            const int rowq = qbase + r + (half << 3);
            const int key0 = kc * 32 + ln;
            const int key1 = key0 + 16;
            float v0 = (key0 <= rowq) ? sacc[0][r] : -1e30f;
            float v1 = (key1 <= rowq) ? sacc[1][r] : -1e30f;
            float mx = fmaxf(v0, v1);
            mx = fmaxf(mx, __shfl_xor(mx, 1, 32));
            mx = fmaxf(mx, __shfl_xor(mx, 2, 32));
            mx = fmaxf(mx, __shfl_xor(mx, 4, 32));
            mx = fmaxf(mx, __shfl_xor(mx, 8, 32));
            float mnew = fmaxf(m[r], mx);
            float p0 = __expf(v0 - mnew);
            float p1 = __expf(v1 - mnew);
            float sum = p0 + p1;
            sum += __shfl_xor(sum, 1, 32);
            sum += __shfl_xor(sum, 2, 32);
            sum += __shfl_xor(sum, 4, 32);
            sum += __shfl_xor(sum, 8, 32);
            float alpha = __expf(m[r] - mnew);
            l[r] = l[r] * alpha + sum;
            m[r] = mnew;
            #pragma unroll
            for (int nt = 0; nt < 8; ++nt)
                acc[nt][r] = acc[nt][r] * alpha;
            Ps[r + (half << 3)][ln]      = (_Float16)p0;
            Ps[r + (half << 3)][16 + ln] = (_Float16)p1;
        }
        __syncthreads();

        // ---- re-load P in A-fragment layout (contiguous 16-half run) ----
        v16h pf;
        {
            int kb = half << 4;
            #pragma unroll
            for (int e = 0; e < 8; ++e) pf[e]     = Ps[ln][kb + e];
            #pragma unroll
            for (int e = 0; e < 8; ++e) pf[8 + e] = Ps[ln][kb + 8 + e];
        }

        // ---- P.V: 8 output-dh tiles of 16 ----
        #pragma unroll
        for (int nt = 0; nt < 8; ++nt) {
            v16h vf;
            int d = nt * 16 + ln;
            int kb = half << 4;                 // contiguous key run
            #pragma unroll
            for (int e = 0; e < 16; ++e) vf[e] = Vt[d][kb + e];
            acc[nt] = wmma_f32_f16(pf, vf, acc[nt]);
        }
        __syncthreads();
    }

    // ---- epilogue: one base pointer, immediate offsets ----
    {
        float* op = attn + ((size_t)(b * SS) + qbase + (half << 3)) * (NQ * DH)
                         + (size_t)hq * DH + ln;
        #pragma unroll
        for (int nt = 0; nt < 8; ++nt)
            #pragma unroll
            for (int r = 0; r < 8; ++r)
                op[(size_t)r * (NQ * DH) + nt * 16] = acc[nt][r] / l[r];
    }
}

// ---------------------------------------------------------------------------
// SwiGLU: act[row,i] = silu(gu[row,i]) * gu[row,I+i]
// ---------------------------------------------------------------------------
__global__ void silu_mul_kernel(const float* __restrict__ gu,
                                float* __restrict__ act) {
    const size_t total = (size_t)BB * SS * II;
    for (size_t idx = (size_t)blockIdx.x * blockDim.x + threadIdx.x; idx < total;
         idx += (size_t)gridDim.x * blockDim.x) {
        size_t row = idx / II;
        int    i   = (int)(idx % II);
        float g = gu[row * (size_t)(2 * II) + i];
        float u = gu[row * (size_t)(2 * II) + II + i];
        act[idx] = (g / (1.f + __expf(-g))) * u;
    }
}

// ---------------------------------------------------------------------------
// Host-side orchestration
// ---------------------------------------------------------------------------
extern "C" void kernel_launch(void* const* d_in, const int* in_sizes, int n_in,
                              void* d_out, int out_size, void* d_ws, size_t ws_size,
                              hipStream_t stream) {
    const float* hidden   = (const float*)d_in[0];
    const float* residual = (const float*)d_in[1];
    const float* cosb     = (const float*)d_in[2];
    const float* sinb     = (const float*)d_in[3];
    const float* w_qkv    = (const float*)d_in[4];
    const float* w_o      = (const float*)d_in[5];
    const float* w_gu     = (const float*)d_in[6];
    const float* w_down   = (const float*)d_in[7];
    const float* g_in     = (const float*)d_in[8];
    const float* g_post   = (const float*)d_in[9];

    float* outp = (float*)d_out;                 // [out | res2]
    const size_t BSH = (size_t)BB * SS * HH;     // 16,777,216
    float* res2 = outp + BSH;

    float* ws    = (float*)d_ws;
    float* res   = ws;  ws += BSH;
    float* h1    = ws;  ws += BSH;
    float* qkv   = ws;  ws += (size_t)BB * SS * QKVD;
    float* attn  = ws;  ws += BSH;
    float* oproj = ws;  ws += BSH;
    float* h2    = ws;  ws += BSH;
    float* gu    = ws;  ws += (size_t)BB * SS * 2 * II;
    float* act   = ws;  ws += (size_t)BB * SS * II;

    const int BS = BB * SS;                      // 4096 rows

    // 1) h1, res = rmsnorm(hidden + residual) * g_in
    rmsnorm_fused_kernel<<<BS, 256, 0, stream>>>(hidden, residual, g_in, h1, res);

    // 2) qkv = h1 @ w_qkv     (4096 x 6144 x 4096)
    gemm_f16_wmma<QKVD, HH><<<(BS / 128) * (QKVD / 128), 256, 0, stream>>>(
        h1, w_qkv, qkv);

    // 3) RoPE on Q and K (in place)
    rope_kernel<<<4096, 256, 0, stream>>>(qkv, cosb, sinb);

    // 4) causal GQA flash attention
    attention_kernel<<<dim3(SS / 16, NQ, BB), 32, 0, stream>>>(qkv, attn);

    // 5) oproj = attn @ w_o   (4096 x 4096 x 4096)
    gemm_f16_wmma<HH, HH><<<(BS / 128) * (HH / 128), 256, 0, stream>>>(
        attn, w_o, oproj);

    // 6) h2, res2 = rmsnorm(oproj + res) * g_post   (res2 -> d_out second half)
    rmsnorm_fused_kernel<<<BS, 256, 0, stream>>>(oproj, res, g_post, h2, res2);

    // 7) gu = h2 @ w_gate_up  (4096 x 28672 x 4096)
    gemm_f16_wmma<2 * II, HH><<<(BS / 128) * ((2 * II) / 128), 256, 0, stream>>>(
        h2, w_gu, gu);

    // 8) act = silu(gate) * up
    silu_mul_kernel<<<8192, 256, 0, stream>>>(gu, act);

    // 9) out = act @ w_down   (4096 x 4096 x 14336)  -> d_out first half
    gemm_f16_wmma<HH, II><<<(BS / 128) * (HH / 128), 256, 0, stream>>>(
        act, w_down, outp);
}
